// PredNet_2834678415760
// MI455X (gfx1250) — matrix operand order repaired and checked by reading, not compiled, not run on hardware
//
#include <hip/hip_runtime.h>

// Problem constants (from reference): B=8, D1=1024, DJ=512, S=2048, OUT=2048, IN=2048
#define B_    8
#define D1_   1024
#define DJ_   512
#define S_    2048
#define OUT_  2048
#define IN_   2048

#define KT        32          // K per WMMA / per LDS stage
#define LDSTRIDE  40          // KT + 8 pad (shorts): 80B rows -> b128-aligned, 64-bank conflict-free
#define MT        128         // block tile along S
#define NT        128         // block tile along OUT

typedef __attribute__((ext_vector_type(16))) __bf16 v16bf;
typedef __attribute__((ext_vector_type(8)))  __bf16 v8bf;
typedef __attribute__((ext_vector_type(8)))  float  v8f;

__device__ __forceinline__ unsigned short bf16_hi_bits(float f) {
    return (unsigned short)(__builtin_bit_cast(unsigned int, f) >> 16);   // truncate to bf16
}
__device__ __forceinline__ float bf16_bits_to_f32(unsigned short h) {
    unsigned int u = ((unsigned int)h) << 16;
    return __builtin_bit_cast(float, u);
}
__device__ __forceinline__ v16bf cat8(v8bf a, v8bf b) {
    return __builtin_shufflevector(a, b, 0,1,2,3,4,5,6,7,8,9,10,11,12,13,14,15);
}

// Pick the concat source row block for feature index k0 (KT divides all boundaries)
__device__ __forceinline__ const float* xsrc_for(int k0, int bz,
                                                 const float* in1, const float* in2a,
                                                 const float* in2b) {
    if (k0 < D1_)        return in1  + ((size_t)bz * D1_ + (size_t)k0)               * S_;
    if (k0 < D1_ + DJ_)  return in2a + ((size_t)bz * DJ_ + (size_t)(k0 - D1_))       * S_;
    return                      in2b + ((size_t)bz * DJ_ + (size_t)(k0 - D1_ - DJ_)) * S_;
}

__global__ __launch_bounds__(256, 2)
void prednet_gemm_bf16x3(const float* __restrict__ in1,
                         const float* __restrict__ in2a,
                         const float* __restrict__ in2b,
                         const float* __restrict__ W,
                         const float* __restrict__ bias,
                         float* __restrict__ out)
{
    // bf16 hi/lo tiles, fragment-layout friendly: A[m][k], B[n][k]
    __shared__ unsigned short As_hi[MT * LDSTRIDE];
    __shared__ unsigned short As_lo[MT * LDSTRIDE];
    __shared__ unsigned short Bs_hi[NT * LDSTRIDE];
    __shared__ unsigned short Bs_lo[NT * LDSTRIDE];

    const int tid  = threadIdx.x;
    const int lane = tid & 31;
    const int wave = tid >> 5;
    const int wm   = wave & 1;    // 2 waves along M: 64 rows each
    const int wn   = wave >> 1;   // 4 waves along N: 32 cols each
    const int ln15 = lane & 15;
    const int lhi  = lane >> 4;   // 0 or 1

    const int o0 = blockIdx.x * NT;
    const int s0 = blockIdx.y * MT;
    const int bz = blockIdx.z;

    // ---- hoisted per-thread staging coordinates ----
    int xfl[4], xsl[4], wol[4], wfl[4];
#pragma unroll
    for (int i = 0; i < 4; ++i) {
        const int idx = tid + 256 * i;    // 0..1023
        xfl[i] = idx >> 5;                // X: k row   0..31
        xsl[i] = (idx & 31) << 2;         // X: m col   0..124 (float4)
        wol[i] = idx >> 3;                // W: n row   0..127
        wfl[i] = (idx & 7) << 2;          // W: k col   0..28  (float4)
    }

    // ---- hoisted per-wave LDS fragment bases (element indices into short arrays) ----
    int afrag[4], bfrag[2];
#pragma unroll
    for (int mi = 0; mi < 4; ++mi)
        afrag[mi] = (wm * 64 + mi * 16 + ln15) * LDSTRIDE + (lhi << 3);
#pragma unroll
    for (int ni = 0; ni < 2; ++ni)
        bfrag[ni] = (wn * 32 + ni * 16 + ln15) * LDSTRIDE + (lhi << 3);

    v8f acc[4][2];
#pragma unroll
    for (int mi = 0; mi < 4; ++mi)
#pragma unroll
        for (int ni = 0; ni < 2; ++ni)
            acc[mi][ni] = (v8f){0.f,0.f,0.f,0.f,0.f,0.f,0.f,0.f};

    // ---- software pipeline: prefetch K-tile 0 into registers ----
    float4 xr[4], wr[4];
    {
        const float* xsrc = xsrc_for(0, bz, in1, in2a, in2b);
#pragma unroll
        for (int i = 0; i < 4; ++i) {
            xr[i] = *reinterpret_cast<const float4*>(xsrc + (size_t)xfl[i] * S_ + s0 + xsl[i]);
            wr[i] = *reinterpret_cast<const float4*>(W + (size_t)(o0 + wol[i]) * IN_ + wfl[i]);
        }
    }

    for (int k0 = 0; k0 < IN_; k0 += KT) {
        __syncthreads();   // all fragment reads of the previous tile are done

        // ---- convert current tile (already in registers) and store to LDS ----
#pragma unroll
        for (int i = 0; i < 4; ++i) {
            float vv[4] = {xr[i].x, xr[i].y, xr[i].z, xr[i].w};
#pragma unroll
            for (int j = 0; j < 4; ++j) {
                unsigned short h = bf16_hi_bits(vv[j]);
                unsigned short l = bf16_hi_bits(vv[j] - bf16_bits_to_f32(h));
                As_hi[(xsl[i] + j) * LDSTRIDE + xfl[i]] = h;   // transposed scatter
                As_lo[(xsl[i] + j) * LDSTRIDE + xfl[i]] = l;
            }
        }
#pragma unroll
        for (int i = 0; i < 4; ++i) {
            float vv[4] = {wr[i].x, wr[i].y, wr[i].z, wr[i].w};
            unsigned short h[4], l[4];
#pragma unroll
            for (int j = 0; j < 4; ++j) {
                h[j] = bf16_hi_bits(vv[j]);
                l[j] = bf16_hi_bits(vv[j] - bf16_bits_to_f32(h[j]));
            }
            unsigned int ph0 = (unsigned int)h[0] | ((unsigned int)h[1] << 16);
            unsigned int ph1 = (unsigned int)h[2] | ((unsigned int)h[3] << 16);
            unsigned int pl0 = (unsigned int)l[0] | ((unsigned int)l[1] << 16);
            unsigned int pl1 = (unsigned int)l[2] | ((unsigned int)l[3] << 16);
            *reinterpret_cast<uint2*>(&Bs_hi[wol[i] * LDSTRIDE + wfl[i]]) = make_uint2(ph0, ph1);
            *reinterpret_cast<uint2*>(&Bs_lo[wol[i] * LDSTRIDE + wfl[i]]) = make_uint2(pl0, pl1);
        }

        // ---- issue next tile's global loads: they retire under the WMMA phase ----
        const int k1 = k0 + KT;
        if (k1 < IN_) {
            const float* xsrc1 = xsrc_for(k1, bz, in1, in2a, in2b);
#pragma unroll
            for (int i = 0; i < 4; ++i) {
                xr[i] = *reinterpret_cast<const float4*>(xsrc1 + (size_t)xfl[i] * S_ + s0 + xsl[i]);
                wr[i] = *reinterpret_cast<const float4*>(W + (size_t)(o0 + wol[i]) * IN_ + k1 + wfl[i]);
            }
        }
        // ---- L2 prefetch two tiles ahead (global_prefetch_b8, no counter traffic) ----
        const int k2 = k0 + 2 * KT;
        if (k2 < IN_) {
            const float* xsrc2 = xsrc_for(k2, bz, in1, in2a, in2b);
            __builtin_prefetch(xsrc2 + (size_t)xfl[0] * S_ + s0 + xsl[0], 0, 1);
            __builtin_prefetch(W + (size_t)(o0 + wol[0]) * IN_ + k2 + wfl[0], 0, 1);
        }

        __syncthreads();   // staged tile visible to all waves

        // ---- B fragments (32x16 bf16): lane holds col n=lane&15, k 8h..8h+7 and 16+8h..16+8h+7 ----
        v16bf bh[2], bl[2];
#pragma unroll
        for (int ni = 0; ni < 2; ++ni) {
            v8bf c0 = *reinterpret_cast<const v8bf*>(&Bs_hi[bfrag[ni]]);
            v8bf c1 = *reinterpret_cast<const v8bf*>(&Bs_hi[bfrag[ni] + 16]);
            bh[ni] = cat8(c0, c1);
            v8bf d0 = *reinterpret_cast<const v8bf*>(&Bs_lo[bfrag[ni]]);
            v8bf d1 = *reinterpret_cast<const v8bf*>(&Bs_lo[bfrag[ni] + 16]);
            bl[ni] = cat8(d0, d1);
        }

        // ---- A fragments (16x32 bf16) + 3 WMMAs per (mi,ni): hi*hi + hi*lo + lo*hi ----
#pragma unroll
        for (int mi = 0; mi < 4; ++mi) {
            v8bf a0 = *reinterpret_cast<const v8bf*>(&As_hi[afrag[mi]]);
            v8bf a1 = *reinterpret_cast<const v8bf*>(&As_hi[afrag[mi] + 16]);  // k 16..23 / 24..31
            v16bf ah = cat8(a0, a1);
            v8bf b0 = *reinterpret_cast<const v8bf*>(&As_lo[afrag[mi]]);
            v8bf b1 = *reinterpret_cast<const v8bf*>(&As_lo[afrag[mi] + 16]);
            v16bf al = cat8(b0, b1);
#pragma unroll
            for (int ni = 0; ni < 2; ++ni) {
                acc[mi][ni] = __builtin_amdgcn_wmma_f32_16x16x32_bf16(
                    false, ah, false, bh[ni], (short)0, acc[mi][ni], false, false);
                acc[mi][ni] = __builtin_amdgcn_wmma_f32_16x16x32_bf16(
                    false, ah, false, bl[ni], (short)0, acc[mi][ni], false, false);
                acc[mi][ni] = __builtin_amdgcn_wmma_f32_16x16x32_bf16(
                    false, al, false, bh[ni], (short)0, acc[mi][ni], false, false);
            }
        }
    }

    // ---- epilogue: D layout -> lane (m = lhi*8 + i, n = lane&15); add bias, store f32 ----
#pragma unroll
    for (int mi = 0; mi < 4; ++mi) {
#pragma unroll
        for (int ni = 0; ni < 2; ++ni) {
            int n  = o0 + wn * 32 + ni * 16 + ln15;
            float bv = bias[n];
            int mrow = s0 + wm * 64 + mi * 16 + (lhi << 3);
            float* op = out + ((size_t)bz * S_ + mrow) * OUT_ + n;
#pragma unroll
            for (int i = 0; i < 8; ++i) {
                op[(size_t)i * OUT_] = acc[mi][ni][i] + bv;
            }
        }
    }
}

extern "C" void kernel_launch(void* const* d_in, const int* in_sizes, int n_in,
                              void* d_out, int out_size, void* d_ws, size_t ws_size,
                              hipStream_t stream) {
    const float* in1  = (const float*)d_in[0];   // [8,1024,2048]
    const float* in2a = (const float*)d_in[1];   // [8,512,2048]
    const float* in2b = (const float*)d_in[2];   // [8,512,2048]
    const float* W    = (const float*)d_in[3];   // [2048,2048]
    const float* bias = (const float*)d_in[4];   // [2048]
    float* out = (float*)d_out;                  // [8,2048,2048]

    dim3 grid(OUT_ / NT, S_ / MT, B_);
    prednet_gemm_bf16x3<<<grid, 256, 0, stream>>>(in1, in2a, in2b, W, bias, out);
}
